// GATNet_16149077033814
// MI455X (gfx1250) — matrix operand order repaired and checked by reading, not compiled, not run on hardware
//
#include <hip/hip_runtime.h>
#include <hip/hip_bf16.h>
#include <math.h>

typedef __attribute__((ext_vector_type(16))) _Float16 v16h;
typedef __attribute__((ext_vector_type(8)))  float    v8f;

#define NNODES 16384
#define NEDGES 131072
#define NB     256
#define FIN    78
#define HIDC   128
#define HEADS  10
#define FCELL  954
#define NOUT   2
#define ETOT   (NEDGES + NNODES)
#define NODES_PER_GRAPH (NNODES / NB)
#define NEG_SLOPE 0.2f

static __device__ __forceinline__ float lrelu_f(float x) { return x > 0.f ? x : NEG_SLOPE * x; }
static __device__ __forceinline__ float elu_f(float x)   { return x > 0.f ? x : (expf(x) - 1.f); }

static __device__ __forceinline__ void atomicMaxF(float* addr, float val) {
  unsigned int* ua = (unsigned int*)addr;
  unsigned int old = *ua;
  while (__uint_as_float(old) < val) {
    unsigned int assumed = old;
    old = atomicCAS(ua, assumed, __float_as_uint(val));
    if (old == assumed) break;
  }
}

// ---------------- fill ----------------
__global__ void fill_kernel(float* __restrict__ p, long long n, float v) {
  long long i = (long long)blockIdx.x * blockDim.x + threadIdx.x;
  if (i < n) p[i] = v;
}

// ---------------- WMMA GEMM: C[M,N] = act(A[M,K] @ B[K,N] + bias[N]) ----------------
// Requirements: M % 16 == 0, N % 32 == 0, K % 2 == 0 (caller guarantees).
// 4 waves / 128-thread block; wave tile 16x32 (two accumulators share one A frag);
// block tile 16x128. Guard-free steady-state K loop (float2 A loads, scalar
// coalesced B loads); single select-guarded K tail; WGP-scope prefetch.
__global__ void wmma_gemm_kernel(const float* __restrict__ A,
                                 const float* __restrict__ Bm,
                                 const float* __restrict__ bias,
                                 float* __restrict__ C,
                                 int M, int N, int K, int act) {
  const int lane = threadIdx.x & 31;
  const int wave = threadIdx.x >> 5;
  const int m0 = blockIdx.x * 16;
  const int n0 = blockIdx.y * 128 + wave * 32;
  if (n0 >= N) return;                       // wave-uniform: EXEC stays all-1s
  const int half = lane >> 4;
  const int mrow = m0 + (lane & 15);
  const int c0 = n0 + (lane & 15);           // subtile 0 column
  const int c1 = c0 + 16;                    // subtile 1 column

  const float* __restrict__ arow = A + (long long)mrow * K;

  v8f acc0 = {}, acc1 = {};
  const int kfull = K & ~31;

  int k0 = 0;
  for (; k0 < kfull; k0 += 32) {
    v16h a, b0v, b1v;
    // A 16x32 f16 frag: lane half 0 -> K 0..15; half 1 -> K 16..31 (matches B)
    // K even + 8-float lane chunks -> float2 (b64) loads are 8B aligned.
    const float2* ap0 = (const float2*)(arow + k0 + half * 16);
    const float2* ap1 = (const float2*)(arow + k0 + half * 16 + 8);
    if (k0 + 32 < kfull) {                   // WGP-scope prefetch of next K tile
      __builtin_prefetch(arow + k0 + 32 + half * 8, 0, 3);
      __builtin_prefetch(Bm + (long long)(k0 + 32 + half * 16) * N + c0, 0, 3);
    }
#pragma unroll
    for (int v = 0; v < 4; ++v) {
      float2 x0 = ap0[v];
      float2 x1 = ap1[v];
      a[2 * v]         = (_Float16)x0.x;
      a[2 * v + 1]     = (_Float16)x0.y;
      a[8 + 2 * v]     = (_Float16)x1.x;
      a[8 + 2 * v + 1] = (_Float16)x1.y;
    }
    // B 32x16 frag: N = lane&15; half 0 -> K 0..15, half 1 -> K 16..31
    const float* bp = Bm + (long long)(k0 + half * 16) * N;
#pragma unroll
    for (int v = 0; v < 8; ++v) {
      b0v[2 * v]     = (_Float16)bp[(long long)(2 * v) * N + c0];
      b0v[2 * v + 1] = (_Float16)bp[(long long)(2 * v + 1) * N + c0];
      b1v[2 * v]     = (_Float16)bp[(long long)(2 * v) * N + c1];
      b1v[2 * v + 1] = (_Float16)bp[(long long)(2 * v + 1) * N + c1];
    }
    acc0 = __builtin_amdgcn_wmma_f32_16x16x32_f16(false, a, false, b0v, (short)0, acc0,
                                                  false, false);
    acc1 = __builtin_amdgcn_wmma_f32_16x16x32_f16(false, a, false, b1v, (short)0, acc1,
                                                  false, false);
  }

  if (k0 < K) {  // K tail: clamped-address loads + selects (no exec-mask branches)
    v16h a, b0v, b1v;
    const int kc = K - 1;
#pragma unroll
    for (int v = 0; v < 8; ++v) {
      int ka = k0 + half * 16 + 2 * v;
      float x0 = arow[min(ka, kc)];
      float x1 = arow[min(ka + 1, kc)];
      a[2 * v]     = (_Float16)((ka     < K) ? x0 : 0.f);
      a[2 * v + 1] = (_Float16)((ka + 1 < K) ? x1 : 0.f);
      int kb = k0 + half * 16 + 2 * v;
      float y0 = Bm[(long long)min(kb, kc) * N + c0];
      float y1 = Bm[(long long)min(kb + 1, kc) * N + c0];
      float y2 = Bm[(long long)min(kb, kc) * N + c1];
      float y3 = Bm[(long long)min(kb + 1, kc) * N + c1];
      b0v[2 * v]     = (_Float16)((kb     < K) ? y0 : 0.f);
      b0v[2 * v + 1] = (_Float16)((kb + 1 < K) ? y1 : 0.f);
      b1v[2 * v]     = (_Float16)((kb     < K) ? y2 : 0.f);
      b1v[2 * v + 1] = (_Float16)((kb + 1 < K) ? y3 : 0.f);
    }
    acc0 = __builtin_amdgcn_wmma_f32_16x16x32_f16(false, a, false, b0v, (short)0, acc0,
                                                  false, false);
    acc1 = __builtin_amdgcn_wmma_f32_16x16x32_f16(false, a, false, b1v, (short)0, acc1,
                                                  false, false);
  }

  const float bv0 = bias ? bias[c0] : 0.f;
  const float bv1 = bias ? bias[c1] : 0.f;
#pragma unroll
  for (int r = 0; r < 8; ++r) {
    int row = m0 + r + 8 * half;             // C/D layout: VGPR r -> M = r (+8 for hi half)
    float v0 = acc0[r] + bv0;
    float v1 = acc1[r] + bv1;
    if (act == 1)      { v0 = fmaxf(v0, 0.f); v1 = fmaxf(v1, 0.f); }
    else if (act == 2) { v0 = elu_f(v0);      v1 = elu_f(v1); }
    C[(long long)row * N + c0] = v0;
    C[(long long)row * N + c1] = v1;
  }
}

// ---------------- tiny output projection: out[256,2] = X[256,128] @ W[128,2] + b ----------------
__global__ void small_out_kernel(const float* __restrict__ X, const float* __restrict__ W,
                                 const float* __restrict__ b, float* __restrict__ out) {
  int t = blockIdx.x * blockDim.x + threadIdx.x;
  if (t >= NB * NOUT) return;
  int row = t / NOUT, col = t % NOUT;
  float s = b[col];
  for (int k = 0; k < HIDC; ++k) s += X[row * HIDC + k] * W[k * NOUT + col];
  out[t] = s;
}

// ---------------- per-(node,head) attention dots ----------------
__global__ void node_alpha_kernel(const float* __restrict__ h,
                                  const float* __restrict__ a_s,
                                  const float* __restrict__ a_d,
                                  float* __restrict__ es, float* __restrict__ ed,
                                  int n, int heads, int ch) {
  int t = blockIdx.x * blockDim.x + threadIdx.x;
  if (t >= n * heads) return;
  int node = t / heads, hh = t % heads;
  const float* hp = h + (long long)node * heads * ch + (long long)hh * ch;
  const float* as = a_s + hh * ch;
  const float* ad = a_d + hh * ch;
  float s = 0.f, d = 0.f;
  for (int c = 0; c < ch; ++c) { float x = hp[c]; s += x * as[c]; d += x * ad[c]; }
  es[t] = s; ed[t] = d;
}

// ---------------- per-(edge,head) segment max ----------------
__global__ void edge_max_kernel(const int* __restrict__ ei,
                                const float* __restrict__ es,
                                const float* __restrict__ ed,
                                float* __restrict__ m, int heads) {
  long long t = (long long)blockIdx.x * blockDim.x + threadIdx.x;
  if (t >= (long long)ETOT * heads) return;
  int e = (int)(t / heads), hh = (int)(t % heads);
  int s = (e < NEDGES) ? ei[e]          : (e - NEDGES);
  int d = (e < NEDGES) ? ei[NEDGES + e] : (e - NEDGES);
  float ev = lrelu_f(es[s * heads + hh] + ed[d * heads + hh]);
  atomicMaxF(&m[d * heads + hh], ev);
}

// ---------------- per-(edge,head) softmax weight + scatter accumulate ----------------
__global__ void edge_accum_kernel(const int* __restrict__ ei,
                                  const float* __restrict__ es,
                                  const float* __restrict__ ed,
                                  const float* __restrict__ m,
                                  const float* __restrict__ h,
                                  float* __restrict__ den, float* __restrict__ agg,
                                  int heads, int ch) {
  long long t = (long long)blockIdx.x * blockDim.x + threadIdx.x;
  if (t >= (long long)ETOT * heads) return;
  int e = (int)(t / heads), hh = (int)(t % heads);
  int s = (e < NEDGES) ? ei[e]          : (e - NEDGES);
  int d = (e < NEDGES) ? ei[NEDGES + e] : (e - NEDGES);
  float ev = lrelu_f(es[s * heads + hh] + ed[d * heads + hh]);
  float w = expf(ev - m[d * heads + hh]);
  atomicAdd(&den[d * heads + hh], w);
  const float* hp = h + (long long)s * heads * ch + (long long)hh * ch;
  float* ap = agg + (long long)d * heads * ch + (long long)hh * ch;
  for (int c = 0; c < ch; ++c) atomicAdd(&ap[c], w * hp[c]);
}

// ---------------- normalize aggregate, add bias, elu (in place) ----------------
__global__ void finalize_kernel(const float* __restrict__ den,
                                const float* __restrict__ bias,
                                float* __restrict__ agg,
                                int n, int heads, int ch) {
  long long t = (long long)blockIdx.x * blockDim.x + threadIdx.x;
  if (t >= (long long)n * heads * ch) return;
  int c = (int)(t % ch);
  int hh = (int)((t / ch) % heads);
  int node = (int)(t / ((long long)ch * heads));
  float v = agg[t] / den[node * heads + hh] + bias[hh * ch + c];
  agg[t] = elu_f(v);
}

// ---------------- global max pool (batch = node / 64, sorted & dense) ----------------
__global__ void pool_max_kernel(const float* __restrict__ x, float* __restrict__ out) {
  int t = blockIdx.x * blockDim.x + threadIdx.x;
  if (t >= NB * HIDC) return;
  int b = t / HIDC, c = t % HIDC;
  float mx = -1e30f;
  for (int i = 0; i < NODES_PER_GRAPH; ++i)
    mx = fmaxf(mx, x[(long long)(b * NODES_PER_GRAPH + i) * HIDC + c]);
  out[t] = mx;
}

// ---------------- row-wise L2 normalize: one block per row ----------------
__global__ void l2norm_kernel(const float* __restrict__ cell, float* __restrict__ out) {
  __shared__ float red[256];
  int b = blockIdx.x;
  int t = threadIdx.x;
  float ss = 0.f;
  for (int c = t; c < FCELL; c += 256) { float v = cell[b * FCELL + c]; ss += v * v; }
  red[t] = ss;
  __syncthreads();
  for (int s = 128; s > 0; s >>= 1) { if (t < s) red[t] += red[t + s]; __syncthreads(); }
  float scale = 1.f / fmaxf(sqrtf(red[0]), 1e-12f);
  for (int c = t; c < FCELL; c += 256) out[b * FCELL + c] = cell[b * FCELL + c] * scale;
}

// ---------------- concat [B,128] and [B,2] -> [B,130] ----------------
__global__ void concat_kernel(const float* __restrict__ g, const float* __restrict__ o,
                              float* __restrict__ out) {
  int t = blockIdx.x * blockDim.x + threadIdx.x;
  if (t >= NB * (HIDC + NOUT)) return;
  int b = t / (HIDC + NOUT), j = t % (HIDC + NOUT);
  out[t] = (j < HIDC) ? g[b * HIDC + j] : o[b * NOUT + (j - HIDC)];
}

// ======================================================================
extern "C" void kernel_launch(void* const* d_in, const int* in_sizes, int n_in,
                              void* d_out, int out_size, void* d_ws, size_t ws_size,
                              hipStream_t stream) {
  (void)in_sizes; (void)n_in; (void)out_size; (void)ws_size;

  const float* x      = (const float*)d_in[0];
  const int*   ei     = (const int*)  d_in[1];
  // d_in[2] (batch) is node/64 sorted-dense; computed arithmetically instead.
  const float* cell   = (const float*)d_in[3];
  const float* W1     = (const float*)d_in[4];
  const float* a_src1 = (const float*)d_in[5];
  const float* a_dst1 = (const float*)d_in[6];
  const float* b1     = (const float*)d_in[7];
  const float* W2     = (const float*)d_in[8];
  const float* a_src2 = (const float*)d_in[9];
  const float* a_dst2 = (const float*)d_in[10];
  const float* b2     = (const float*)d_in[11];
  const float* Wg     = (const float*)d_in[12];
  const float* bg     = (const float*)d_in[13];
  const float* Wf1    = (const float*)d_in[14];
  const float* bf1    = (const float*)d_in[15];
  const float* Wf2    = (const float*)d_in[16];
  const float* bf2    = (const float*)d_in[17];
  const float* Wf3    = (const float*)d_in[18];
  const float* bf3    = (const float*)d_in[19];
  const float* Wo     = (const float*)d_in[20];
  const float* bo     = (const float*)d_in[21];

  // ---- workspace carve (aligned 256B) ----
  char* wsp = (char*)d_ws;
  auto carve = [&](size_t elems) -> float* {
    float* p = (float*)wsp;
    size_t bytes = (elems * sizeof(float) + 255) & ~(size_t)255;
    wsp += bytes;
    return p;
  };
  float* h1   = carve((size_t)NNODES * HEADS * HIDC);   // 80 MB
  float* agg1 = carve((size_t)NNODES * HEADS * HIDC);   // 80 MB -> becomes x1
  float* es1  = carve((size_t)NNODES * HEADS);
  float* ed1  = carve((size_t)NNODES * HEADS);
  float* m1   = carve((size_t)NNODES * HEADS);
  float* den1 = carve((size_t)NNODES * HEADS);
  float* h2   = carve((size_t)NNODES * HIDC);           // 8 MB
  float* agg2 = carve((size_t)NNODES * HIDC);           // 8 MB -> becomes x2
  float* es2  = carve((size_t)NNODES);
  float* ed2  = carve((size_t)NNODES);
  float* m2   = carve((size_t)NNODES);
  float* den2 = carve((size_t)NNODES);
  float* pool = carve((size_t)NB * HIDC);
  float* gfc  = carve((size_t)NB * HIDC);
  float* celln= carve((size_t)NB * FCELL);
  float* c1   = carve((size_t)NB * 2048);
  float* c2   = carve((size_t)NB * 512);
  float* c3   = carve((size_t)NB * HIDC);
  float* ofin = carve((size_t)NB * NOUT);

  auto cdiv = [](long long a, long long b) -> unsigned { return (unsigned)((a + b - 1) / b); };
  auto fill = [&](float* p, long long n, float v) {
    fill_kernel<<<cdiv(n, 256), 256, 0, stream>>>(p, n, v);
  };
  auto gemm = [&](const float* A, const float* B, const float* bias, float* C,
                  int M, int N, int K, int act) {
    dim3 grid(cdiv(M, 16), cdiv(N, 128));
    wmma_gemm_kernel<<<grid, 128, 0, stream>>>(A, B, bias, C, M, N, K, act);
  };

  // ===== GAT layer 1 =====
  gemm(x, W1, nullptr, h1, NNODES, HEADS * HIDC, FIN, 0);           // h1 = x @ W1
  fill(m1,   (long long)NNODES * HEADS, -1e30f);
  fill(den1, (long long)NNODES * HEADS, 0.f);
  fill(agg1, (long long)NNODES * HEADS * HIDC, 0.f);
  node_alpha_kernel<<<cdiv((long long)NNODES * HEADS, 256), 256, 0, stream>>>(
      h1, a_src1, a_dst1, es1, ed1, NNODES, HEADS, HIDC);
  edge_max_kernel<<<cdiv((long long)ETOT * HEADS, 256), 256, 0, stream>>>(
      ei, es1, ed1, m1, HEADS);
  edge_accum_kernel<<<cdiv((long long)ETOT * HEADS, 256), 256, 0, stream>>>(
      ei, es1, ed1, m1, h1, den1, agg1, HEADS, HIDC);
  finalize_kernel<<<cdiv((long long)NNODES * HEADS * HIDC, 256), 256, 0, stream>>>(
      den1, b1, agg1, NNODES, HEADS, HIDC);                         // agg1 = x1 (elu'd)

  // ===== GAT layer 2 =====
  gemm(agg1, W2, nullptr, h2, NNODES, HIDC, HEADS * HIDC, 0);       // h2 = x1 @ W2
  fill(m2,   (long long)NNODES, -1e30f);
  fill(den2, (long long)NNODES, 0.f);
  fill(agg2, (long long)NNODES * HIDC, 0.f);
  node_alpha_kernel<<<cdiv((long long)NNODES, 256), 256, 0, stream>>>(
      h2, a_src2, a_dst2, es2, ed2, NNODES, 1, HIDC);
  edge_max_kernel<<<cdiv((long long)ETOT, 256), 256, 0, stream>>>(
      ei, es2, ed2, m2, 1);
  edge_accum_kernel<<<cdiv((long long)ETOT, 256), 256, 0, stream>>>(
      ei, es2, ed2, m2, h2, den2, agg2, 1, HIDC);
  finalize_kernel<<<cdiv((long long)NNODES * HIDC, 256), 256, 0, stream>>>(
      den2, b2, agg2, NNODES, 1, HIDC);                             // agg2 = x2 (elu'd)

  // ===== pool + graph FC =====
  pool_max_kernel<<<cdiv((long long)NB * HIDC, 256), 256, 0, stream>>>(agg2, pool);
  gemm(pool, Wg, bg, gfc, NB, HIDC, HIDC, 1);                       // relu

  // ===== cell branch =====
  l2norm_kernel<<<NB, 256, 0, stream>>>(cell, celln);
  gemm(celln, Wf1, bf1, c1, NB, 2048, FCELL, 1);
  gemm(c1, Wf2, bf2, c2, NB, 512, 2048, 1);
  gemm(c2, Wf3, bf3, c3, NB, HIDC, 512, 1);
  small_out_kernel<<<cdiv((long long)NB * NOUT, 256), 256, 0, stream>>>(c3, Wo, bo, ofin);

  // ===== concat output [256,130] =====
  concat_kernel<<<cdiv((long long)NB * (HIDC + NOUT), 256), 256, 0, stream>>>(
      gfc, ofin, (float*)d_out);
}
